// RNNDetectorStage_60361470378303
// MI455X (gfx1250) — matrix operand order repaired and checked by reading, not compiled, not run on hardware
//
#include <hip/hip_runtime.h>
#include <math.h>

// ---------------- problem constants ----------------
#define L_SEQ 8
#define BATCH 4
#define C_IN  32
#define H_IN  96
#define W_IN  128
#define CCH   64          // stage channel dim C
#define DI    128         // d_inner
#define NST   8           // d_state
#define RDT   4           // dt rank
#define H2    48
#define W2    64
#define LB    (L_SEQ*BATCH)   // 32
#define HW2   (H2*W2)         // 3072
#define TOK   98304           // tokens per stage (same for all 3 stages)

typedef __attribute__((ext_vector_type(16))) _Float16 v16h;
typedef __attribute__((ext_vector_type(8)))  _Float16 v8h;
typedef __attribute__((ext_vector_type(8)))  float    v8f;

static inline int cdiv_i(int a, int b) { return (a + b - 1) / b; }

__device__ __forceinline__ float dev_sigmoid(float x) { return 1.f / (1.f + expf(-x)); }
__device__ __forceinline__ float dev_silu(float x)    { return x / (1.f + expf(-x)); }
__device__ __forceinline__ float dev_gelu_tanh(float x) {
  float x3 = x * x * x;
  return 0.5f * x * (1.f + tanhf(0.7978845608028654f * (x + 0.044715f * x3)));
}

// =====================================================================
// Weight prep: f32 (K,N) row-major -> f16 (N,K) "transposed" so each
// lane's B fragment (k = k0 + 16*half + j, j=0..15) is one contiguous
// 32-byte v16h load.
// =====================================================================
__global__ void k_w2f16t(const float* __restrict__ W, _Float16* __restrict__ Bt,
                         int K, int N) {
  int idx = blockIdx.x * blockDim.x + threadIdx.x;
  if (idx >= K * N) return;
  int n = idx / K, k = idx % K;
  Bt[idx] = (_Float16)W[(size_t)k * N + n];
}

// =====================================================================
// WMMA GEMM, f16 in / f32 out: C[M,N] = A[M,K] * Bt[N,K]^T.
// One wave computes a 16x64 strip (4 accumulators, 4 WMMAs per K-step),
// reusing the A fragment 4x. All fragment loads are contiguous vector
// loads in this code's contiguous k order (A and B alike):
//   A (16x32): lane lid = M row; k in [16h, 16h+16)
//   B (32x16): lane lid = N col; k in [16h, 16h+16)  (contiguous in N-major Bt)
//   C/D: vgpr r -> row 8*half+r, col lid
// M%16==0, N%64==0, K%32==0. Wave-uniform early-out keeps EXEC all-1.
// =====================================================================
__global__ void k_gemm_wmma_strip(const _Float16* __restrict__ A,
                                  const _Float16* __restrict__ Bt,
                                  float* __restrict__ C, int M, int N, int K) {
  const int lane = threadIdx.x & 31;
  const int wave = (blockIdx.x * blockDim.x + threadIdx.x) >> 5;
  const int sT = N >> 6;                   // 64-wide strips
  if (wave >= (M >> 4) * sT) return;
  const int m0 = (wave / sT) << 4;
  const int n0 = (wave % sT) << 6;
  const int half = lane >> 4;
  const int lid  = lane & 15;

  const _Float16* __restrict__ Arow = A + (size_t)(m0 + lid) * K;
  v8f acc0 = {}, acc1 = {}, acc2 = {}, acc3 = {};
  for (int k0 = 0; k0 < K; k0 += 32) {
    v16h af;
    ((v8h*)&af)[0] = *(const v8h*)(Arow + k0 + half * 16);
    ((v8h*)&af)[1] = *(const v8h*)(Arow + k0 + half * 16 + 8);
    const _Float16* __restrict__ Bb = Bt + (size_t)(n0 + lid) * K + k0 + half * 16;
    v16h b0 = *(const v16h*)(Bb);
    v16h b1 = *(const v16h*)(Bb + (size_t)16 * K);
    v16h b2 = *(const v16h*)(Bb + (size_t)32 * K);
    v16h b3 = *(const v16h*)(Bb + (size_t)48 * K);
    acc0 = __builtin_amdgcn_wmma_f32_16x16x32_f16(false, af, false, b0, (short)0, acc0, false, false);
    acc1 = __builtin_amdgcn_wmma_f32_16x16x32_f16(false, af, false, b1, (short)0, acc1, false, false);
    acc2 = __builtin_amdgcn_wmma_f32_16x16x32_f16(false, af, false, b2, (short)0, acc2, false, false);
    acc3 = __builtin_amdgcn_wmma_f32_16x16x32_f16(false, af, false, b3, (short)0, acc3, false, false);
  }
#pragma unroll
  for (int r = 0; r < 8; ++r) {
    float* crow = C + (size_t)(m0 + half * 8 + r) * N + n0 + lid;
    crow[0]  = acc0[r];
    crow[16] = acc1[r];
    crow[32] = acc2[r];
    crow[48] = acc3[r];
  }
}

// =====================================================================
// Downsample conv 3x3 stride 2 SAME (pad lo=0, hi=1), 32 -> 64 ch.
// x: (LB, C_IN, 96, 128)  ->  xf: (LB, 64, 48, 64)
// =====================================================================
__global__ void k_dsconv(const float* __restrict__ x, const float* __restrict__ w,
                         const float* __restrict__ b, float* __restrict__ xf) {
  int idx = blockIdx.x * blockDim.x + threadIdx.x;
  if (idx >= LB * CCH * HW2) return;
  int ow = idx % W2;  int t = idx / W2;
  int oh = t % H2;    t /= H2;
  int oc = t % CCH;   int lb = t / CCH;
  float acc = b[oc];
  for (int kh = 0; kh < 3; ++kh) {
    int ih = oh * 2 + kh;
    if (ih >= H_IN) continue;
    for (int kw = 0; kw < 3; ++kw) {
      int iw = ow * 2 + kw;
      if (iw >= W_IN) continue;
      const float* xp = x + ((size_t)(lb * C_IN) * H_IN + ih) * W_IN + iw;
      const float* wp = w + ((size_t)(oc * C_IN) * 3 + kh) * 3 + kw;
      for (int ic = 0; ic < C_IN; ++ic)
        acc += xp[(size_t)ic * H_IN * W_IN] * wp[(size_t)ic * 9];
    }
  }
  xf[idx] = acc;
}

// =====================================================================
// Permute (stage-dependent gather) + LayerNorm over C=64.
// Writes xn (f32, for residual) and xn_h (f16, GEMM A operand).
// stage 0: src = xf (LB,C,H2,W2);  token = (b*W2+w)*384 + l*48 + h
// stage 1: src = prev stage tokens; token = (b*H2+h)*512 + l*W2 + w
// stage 2: src = prev stage tokens; token = ((l*B+b)*H2+h)*W2 + w
// =====================================================================
__global__ void k_perm_ln(const float* __restrict__ src, float* __restrict__ xn,
                          _Float16* __restrict__ xn_h,
                          const float* __restrict__ g, const float* __restrict__ beta,
                          int stage) {
  int tkn = blockIdx.x * blockDim.x + threadIdx.x;
  if (tkn >= TOK) return;
  size_t base; int cstride;
  if (stage == 0) {
    int bw = tkn / 384, pos = tkn % 384;
    int l = pos / 48, h = pos % 48;
    int b = bw / W2,  w = bw % W2;
    base = ((size_t)(l * BATCH + b) * CCH) * HW2 + (size_t)h * W2 + w;
    cstride = HW2;
  } else if (stage == 1) {
    int bh = tkn / 512, pos = tkn % 512;
    int l = pos / W2, w = pos % W2;
    int b = bh / H2,  h = bh % H2;
    int st = (b * W2 + w) * 384 + l * 48 + h;
    base = (size_t)st * CCH; cstride = 1;
  } else {
    int lb = tkn / HW2, hw = tkn % HW2;
    int h = hw / W2, w = hw % W2;
    int l = lb / BATCH, b = lb % BATCH;
    int st = (b * H2 + h) * 512 + l * W2 + w;
    base = (size_t)st * CCH; cstride = 1;
  }
  float mean = 0.f;
  for (int c = 0; c < CCH; ++c) mean += src[base + (size_t)c * cstride];
  mean *= (1.f / CCH);
  float var = 0.f;
  for (int c = 0; c < CCH; ++c) {
    float d = src[base + (size_t)c * cstride] - mean;
    var += d * d;
  }
  var *= (1.f / CCH);
  float r = rsqrtf(var + 1e-5f);
  float* o = xn + (size_t)tkn * CCH;
  _Float16* oh = xn_h + (size_t)tkn * CCH;
  for (int c = 0; c < CCH; ++c) {
    float v = (src[base + (size_t)c * cstride] - mean) * r * g[c] + beta[c];
    o[c] = v;
    oh[c] = (_Float16)v;
  }
}

// =====================================================================
// Depthwise 3x3 SAME (pad 1) + SiLU on xc (= first 128 cols of xz).
// xz: (TOK, 256) over spatial (Bb, Hs, Ws); u: (TOK, 128)
// =====================================================================
__global__ void k_dwconv_silu(const float* __restrict__ xz, const float* __restrict__ w,
                              const float* __restrict__ bias, float* __restrict__ u,
                              int Hs, int Ws) {
  int idx = blockIdx.x * blockDim.x + threadIdx.x;
  if (idx >= TOK * DI) return;
  int d = idx % DI; int tkn = idx / DI;
  int HWs = Hs * Ws;
  int pos = tkn % HWs; int bb = tkn / HWs;
  int h = pos / Ws, ww = pos % Ws;
  float acc = bias[d];
  for (int kh = 0; kh < 3; ++kh) {
    int ih = h + kh - 1;
    if (ih < 0 || ih >= Hs) continue;
    for (int kw = 0; kw < 3; ++kw) {
      int iw = ww + kw - 1;
      if (iw < 0 || iw >= Ws) continue;
      int nt = (bb * Hs + ih) * Ws + iw;
      acc += xz[(size_t)nt * 256 + d] * w[d * 9 + kh * 3 + kw];
    }
  }
  u[(size_t)tkn * DI + d] = dev_silu(acc);
}

// x-proj: xbc[TOK,24] = u[TOK,128] @ wx[128,24]
__global__ void k_xproj(const float* __restrict__ u, const float* __restrict__ wx,
                        float* __restrict__ xbc) {
  int idx = blockIdx.x * blockDim.x + threadIdx.x;
  if (idx >= TOK * 24) return;
  int j = idx % 24; int tkn = idx / 24;
  const float* ur = u + (size_t)tkn * DI;
  float acc = 0.f;
  for (int k = 0; k < DI; ++k) acc += ur[k] * wx[k * 24 + j];
  xbc[idx] = acc;
}

// dt = softplus(dtl @ wdt + bdt); dtl = xbc[:, 0:4]
__global__ void k_dtproj(const float* __restrict__ xbc, const float* __restrict__ wdt,
                         const float* __restrict__ bdt, float* __restrict__ dt) {
  int idx = blockIdx.x * blockDim.x + threadIdx.x;
  if (idx >= TOK * DI) return;
  int d = idx % DI; int tkn = idx / DI;
  float acc = bdt[d];
  const float* xr = xbc + (size_t)tkn * 24;
  for (int r = 0; r < RDT; ++r) acc += xr[r] * wdt[r * DI + d];
  dt[idx] = (acc > 20.f) ? acc : log1pf(expf(acc));
}

// =====================================================================
// Selective scan: one thread per (bb, d) recurrence, 8 states in regs.
// Latency-bound serial chain -> prefetch next timestep's rows
// (global_prefetch_b8) while the recurrence FMAs execute.
// =====================================================================
__global__ void k_scan(const float* __restrict__ dt, const float* __restrict__ u,
                       const float* __restrict__ xbc, const float* __restrict__ alog,
                       const float* __restrict__ dpar, float* __restrict__ y,
                       int Bb, int Lscan) {
  int idx = blockIdx.x * blockDim.x + threadIdx.x;
  if (idx >= Bb * DI) return;
  int d = idx % DI; int bb = idx / DI;
  float A[NST], h[NST];
#pragma unroll
  for (int n = 0; n < NST; ++n) { A[n] = -expf(alog[d * NST + n]); h[n] = 0.f; }
  float dp = dpar[d];
  size_t tkn = (size_t)bb * Lscan;
  for (int t = 0; t < Lscan; ++t, ++tkn) {
    if (t + 1 < Lscan) {
      __builtin_prefetch(dt + (tkn + 1) * DI + d, 0, 3);
      __builtin_prefetch(u + (tkn + 1) * DI + d, 0, 3);
      __builtin_prefetch(xbc + (tkn + 1) * 24, 0, 3);
    }
    float dtv = dt[tkn * DI + d];
    float uv  = u[tkn * DI + d];
    float du  = dtv * uv;
    const float* bc = xbc + tkn * 24;
    float yv = 0.f;
#pragma unroll
    for (int n = 0; n < NST; ++n) {
      h[n] = expf(dtv * A[n]) * h[n] + du * bc[4 + n];
      yv += h[n] * bc[12 + n];
    }
    y[tkn * DI + d] = yv + uv * dp;
  }
}

// gate + LN over DI: yg_h = f16( LN(y)*silu(z) );  z = xz[:, 128:256]
__global__ void k_gate_ln(const float* __restrict__ y, const float* __restrict__ xz,
                          const float* __restrict__ g, const float* __restrict__ beta,
                          _Float16* __restrict__ yg_h) {
  int tkn = blockIdx.x * blockDim.x + threadIdx.x;
  if (tkn >= TOK) return;
  const float* yr = y + (size_t)tkn * DI;
  float mean = 0.f;
  for (int d = 0; d < DI; ++d) mean += yr[d];
  mean *= (1.f / DI);
  float var = 0.f;
  for (int d = 0; d < DI; ++d) { float e = yr[d] - mean; var += e * e; }
  var *= (1.f / DI);
  float r = rsqrtf(var + 1e-5f);
  const float* zr = xz + (size_t)tkn * 256 + DI;
  _Float16* o = yg_h + (size_t)tkn * DI;
  for (int d = 0; d < DI; ++d) {
    float yn = (yr[d] - mean) * r * g[d] + beta[d];
    o[d] = (_Float16)(yn * dev_silu(zr[d]));
  }
}

// CAB conv1: 3x3 SAME, 64->16, gelu(tanh). yo: (TOK,64) spatial (Bb,Hs,Ws)
__global__ void k_cab_conv1(const float* __restrict__ yo, const float* __restrict__ w1,
                            const float* __restrict__ b1, float* __restrict__ c1,
                            int Hs, int Ws) {
  int idx = blockIdx.x * blockDim.x + threadIdx.x;
  if (idx >= TOK * 16) return;
  int oc = idx % 16; int tkn = idx / 16;
  int HWs = Hs * Ws;
  int pos = tkn % HWs; int bb = tkn / HWs;
  int h = pos / Ws, ww = pos % Ws;
  float acc = b1[oc];
  for (int kh = 0; kh < 3; ++kh) {
    int ih = h + kh - 1;
    if (ih < 0 || ih >= Hs) continue;
    for (int kw = 0; kw < 3; ++kw) {
      int iw = ww + kw - 1;
      if (iw < 0 || iw >= Ws) continue;
      const float* yp = yo + (size_t)((bb * Hs + ih) * Ws + iw) * CCH;
      const float* wp = w1 + ((size_t)oc * CCH * 9) + kh * 3 + kw;
      for (int ic = 0; ic < CCH; ++ic) acc += yp[ic] * wp[(size_t)ic * 9];
    }
  }
  c1[(size_t)tkn * 16 + oc] = dev_gelu_tanh(acc);
}

// CAB conv2: 3x3 SAME, 16->64
__global__ void k_cab_conv2(const float* __restrict__ c1, const float* __restrict__ w2,
                            const float* __restrict__ b2, float* __restrict__ c2,
                            int Hs, int Ws) {
  int idx = blockIdx.x * blockDim.x + threadIdx.x;
  if (idx >= TOK * CCH) return;
  int oc = idx % CCH; int tkn = idx / CCH;
  int HWs = Hs * Ws;
  int pos = tkn % HWs; int bb = tkn / HWs;
  int h = pos / Ws, ww = pos % Ws;
  float acc = b2[oc];
  for (int kh = 0; kh < 3; ++kh) {
    int ih = h + kh - 1;
    if (ih < 0 || ih >= Hs) continue;
    for (int kw = 0; kw < 3; ++kw) {
      int iw = ww + kw - 1;
      if (iw < 0 || iw >= Ws) continue;
      const float* cp = c1 + (size_t)((bb * Hs + ih) * Ws + iw) * 16;
      const float* wp = w2 + ((size_t)oc * 16 * 9) + kh * 3 + kw;
      for (int ic = 0; ic < 16; ++ic) acc += cp[ic] * wp[(size_t)ic * 9];
    }
  }
  c2[idx] = acc;
}

// channel squeeze: mean over (Hs,Ws) per (bb, c). one block per (bb,c).
__global__ void k_squeeze(const float* __restrict__ c2, float* __restrict__ s,
                          int Hs, int Ws) {
  int bc = blockIdx.x;
  int c = bc % CCH; int bb = bc / CCH;
  int HWs = Hs * Ws;
  float acc = 0.f;
  for (int i = threadIdx.x; i < HWs; i += blockDim.x)
    acc += c2[((size_t)bb * HWs + i) * CCH + c];
  __shared__ float red[256];
  red[threadIdx.x] = acc;
  __syncthreads();
  for (int st = 128; st > 0; st >>= 1) {
    if ((int)threadIdx.x < st) red[threadIdx.x] += red[threadIdx.x + st];
    __syncthreads();
  }
  if (threadIdx.x == 0) s[bc] = red[0] / (float)HWs;
}

// channel attention MLP: sig[bb,c] = sigmoid( relu(s@w1^T+b1) @ w2^T + b2 )
__global__ void k_ca_mlp(const float* __restrict__ s, const float* __restrict__ w1,
                         const float* __restrict__ b1, const float* __restrict__ w2,
                         const float* __restrict__ b2, float* __restrict__ sig, int Bb) {
  int idx = blockIdx.x * blockDim.x + threadIdx.x;
  if (idx >= Bb * CCH) return;
  int c = idx % CCH; int bb = idx / CCH;
  const float* sr = s + (size_t)bb * CCH;
  float acc = b2[c];
  for (int j = 0; j < 4; ++j) {
    float s1 = b1[j];
    for (int k = 0; k < CCH; ++k) s1 += sr[k] * w1[j * CCH + k];
    s1 = fmaxf(s1, 0.f);
    acc += s1 * w2[c * 4 + j];
  }
  sig[idx] = dev_sigmoid(acc);
}

// final: tout = c2 * sig[bb,c] + xn * skip[c]
__global__ void k_scale_add(const float* __restrict__ c2, const float* __restrict__ sig,
                            const float* __restrict__ xn, const float* __restrict__ skip,
                            float* __restrict__ tout, int HWs) {
  int idx = blockIdx.x * blockDim.x + threadIdx.x;
  if (idx >= TOK * CCH) return;
  int c = idx % CCH; int tkn = idx / CCH;
  int bb = tkn / HWs;
  tout[idx] = c2[idx] * sig[bb * CCH + c] + xn[idx] * skip[c];
}

// final permute: t3 tokens -> output (L,B,C,H2,W2)
__global__ void k_out_perm(const float* __restrict__ t3, float* __restrict__ out) {
  int idx = blockIdx.x * blockDim.x + threadIdx.x;
  if (idx >= LB * CCH * HW2) return;
  int w = idx % W2;  int t = idx / W2;
  int h = t % H2;    t /= H2;
  int c = t % CCH;   int lb = t / CCH;
  int tkn = (lb * H2 + h) * W2 + w;
  out[idx] = t3[(size_t)tkn * CCH + c];
}

// =====================================================================
// host side
// =====================================================================
extern "C" void kernel_launch(void* const* d_in, const int* in_sizes, int n_in,
                              void* d_out, int out_size, void* d_ws, size_t ws_size,
                              hipStream_t stream) {
  (void)in_sizes; (void)n_in; (void)out_size; (void)ws_size;
  const float* x       = (const float*)d_in[0];
  const float* ds_w    = (const float*)d_in[1];
  const float* ds_b    = (const float*)d_in[2];
  const float* ln_g    = (const float*)d_in[3];
  const float* ln_b    = (const float*)d_in[4];
  const float* skip    = (const float*)d_in[5];
  const float* ss_win  = (const float*)d_in[6];   // (3,64,256)
  const float* ss_cw   = (const float*)d_in[7];   // (3,128,1,3,3)
  const float* ss_cb   = (const float*)d_in[8];
  const float* ss_wx   = (const float*)d_in[9];   // (3,128,24)
  const float* ss_wdt  = (const float*)d_in[10];  // (3,4,128)
  const float* ss_bdt  = (const float*)d_in[11];
  const float* ss_alog = (const float*)d_in[12];  // (3,128,8)
  const float* ss_d    = (const float*)d_in[13];
  const float* ss_lng  = (const float*)d_in[14];
  const float* ss_lnb  = (const float*)d_in[15];
  const float* ss_wout = (const float*)d_in[16];  // (3,128,64)
  const float* cab_w1  = (const float*)d_in[17];  // (3,16,64,3,3)
  const float* cab_b1  = (const float*)d_in[18];
  const float* cab_w2  = (const float*)d_in[19];  // (3,64,16,3,3)
  const float* cab_b2  = (const float*)d_in[20];
  const float* ca_w1   = (const float*)d_in[21];  // (3,4,64)
  const float* ca_b1   = (const float*)d_in[22];
  const float* ca_w2   = (const float*)d_in[23];  // (3,64,4)
  const float* ca_b2   = (const float*)d_in[24];

  float* ws = (float*)d_ws;
  const size_t N64  = (size_t)TOK * 64;
  const size_t N128 = (size_t)TOK * 128;
  float* xf   = ws;                        // N64
  float* tA   = xf   + N64;                // N64
  float* tB   = tA   + N64;                // N64
  float* xn   = tB   + N64;                // N64
  float* xz   = xn   + N64;                // TOK*256
  float* u    = xz   + (size_t)TOK * 256;  // N128
  float* xbc  = u    + N128;               // TOK*24
  float* dt   = xbc  + (size_t)TOK * 24;   // N128
  float* ysc  = dt   + N128;               // N128
  float* yo   = ysc  + N128;               // N64
  float* c1   = yo   + N64;                // TOK*16
  float* c2   = c1   + (size_t)TOK * 16;   // N64
  float* sq   = c2   + N64;                // <= 256*64
  float* sig  = sq   + 256 * 64;           // <= 256*64
  _Float16* xn_h  = (_Float16*)(sig + 256 * 64);      // TOK*64 halves
  _Float16* yg_h  = xn_h + N64;                       // TOK*128 halves
  _Float16* winT  = yg_h + N128;                      // 256*64 halves
  _Float16* woutT = winT + 256 * 64;                  // 64*128 halves

  const int TB = 256;
  // stage geometry: (Bb, Hs, Ws)
  const int BbA[3] = {256, 192, 32};
  const int HsA[3] = {8, 8, 48};
  const int WsA[3] = {48, 64, 64};

  k_dsconv<<<cdiv_i(LB * CCH * HW2, TB), TB, 0, stream>>>(x, ds_w, ds_b, xf);

  const float* src = xf;
  float* dst = tA;
  for (int i = 0; i < 3; ++i) {
    const int Bb = BbA[i], Hs = HsA[i], Wsz = WsA[i];
    const int Lscan = Hs * Wsz;

    // weight prep (tiny): f16 N-major copies of win (64x256), wout (128x64)
    k_w2f16t<<<cdiv_i(64 * 256, TB), TB, 0, stream>>>(ss_win + (size_t)i * 64 * 256, winT, 64, 256);
    k_w2f16t<<<cdiv_i(128 * 64, TB), TB, 0, stream>>>(ss_wout + (size_t)i * 128 * 64, woutT, 128, 64);

    k_perm_ln<<<cdiv_i(TOK, TB), TB, 0, stream>>>(src, xn, xn_h,
                                                  ln_g + i * 64, ln_b + i * 64, i);

    // in-proj GEMM: (TOK,64) x (64,256) -> xz  [WMMA, 16x64 strips]
    {
      int waves = (TOK / 16) * (256 / 64);
      k_gemm_wmma_strip<<<cdiv_i(waves * 32, TB), TB, 0, stream>>>(
          xn_h, winT, xz, TOK, 256, 64);
    }

    k_dwconv_silu<<<cdiv_i(TOK * DI, TB), TB, 0, stream>>>(
        xz, ss_cw + (size_t)i * DI * 9, ss_cb + i * DI, u, Hs, Wsz);

    k_xproj<<<cdiv_i(TOK * 24, TB), TB, 0, stream>>>(u, ss_wx + (size_t)i * DI * 24, xbc);

    k_dtproj<<<cdiv_i(TOK * DI, TB), TB, 0, stream>>>(
        xbc, ss_wdt + (size_t)i * RDT * DI, ss_bdt + i * DI, dt);

    k_scan<<<cdiv_i(Bb * DI, TB), TB, 0, stream>>>(
        dt, u, xbc, ss_alog + (size_t)i * DI * NST, ss_d + i * DI, ysc, Bb, Lscan);

    k_gate_ln<<<cdiv_i(TOK, TB), TB, 0, stream>>>(ysc, xz, ss_lng + i * DI,
                                                  ss_lnb + i * DI, yg_h);

    // out-proj GEMM: (TOK,128) x (128,64) -> yo  [WMMA, 16x64 strips]
    {
      int waves = (TOK / 16) * (64 / 64);
      k_gemm_wmma_strip<<<cdiv_i(waves * 32, TB), TB, 0, stream>>>(
          yg_h, woutT, yo, TOK, 64, DI);
    }

    k_cab_conv1<<<cdiv_i(TOK * 16, TB), TB, 0, stream>>>(
        yo, cab_w1 + (size_t)i * 16 * 64 * 9, cab_b1 + i * 16, c1, Hs, Wsz);
    k_cab_conv2<<<cdiv_i(TOK * CCH, TB), TB, 0, stream>>>(
        c1, cab_w2 + (size_t)i * 64 * 16 * 9, cab_b2 + i * 64, c2, Hs, Wsz);
    k_squeeze<<<Bb * CCH, 256, 0, stream>>>(c2, sq, Hs, Wsz);
    k_ca_mlp<<<cdiv_i(Bb * CCH, TB), TB, 0, stream>>>(
        sq, ca_w1 + (size_t)i * 4 * 64, ca_b1 + i * 4,
        ca_w2 + (size_t)i * 64 * 4, ca_b2 + i * 64, sig, Bb);
    k_scale_add<<<cdiv_i(TOK * CCH, TB), TB, 0, stream>>>(
        c2, sig, xn, skip + i * 64, dst, Hs * Wsz);

    src = dst;
    dst = (dst == tA) ? tB : tA;
  }

  k_out_perm<<<cdiv_i(LB * CCH * HW2, TB), TB, 0, stream>>>(src, (float*)d_out);
}
